// SDT_7687991460616
// MI455X (gfx1250) — matrix operand order repaired and checked by reading, not compiled, not run on hardware
//
#include <hip/hip_runtime.h>

#define BATCH   32768
#define INPUT_D 1024
#define NINT    255
#define NINTP   256
#define WIROW   1025
#define NLEAF   256
#define NOUT    1000

typedef __attribute__((ext_vector_type(16))) __bf16 v16bf;
typedef __attribute__((ext_vector_type(8)))  __bf16 v8bf;
typedef __attribute__((ext_vector_type(8)))  float  v8f;

static __device__ __forceinline__ v16bf cat8(v8bf lo, v8bf hi) {
    return __builtin_shufflevector(lo, hi, 0,1,2,3,4,5,6,7,8,9,10,11,12,13,14,15);
}

// ---------------------------------------------------------------------------
// Kernel 1: P[b,n] = sigmoid(W_inner[n,0] + sum_k x[b,k]*W_inner[n,1+k])
// BM=128 x N=256(pad), K=1024 in 32 steps of 32.
// Double-buffered LDS pipeline: global loads for tile kk+1 are issued into
// registers BEFORE the 16 WMMAs on tile kk, so HBM/L2 latency hides behind
// the matrix burst; one barrier per K-step.
// ---------------------------------------------------------------------------
__global__ __launch_bounds__(256, 1)
void sdt_gemm1_sigmoid(const float* __restrict__ x,
                       const float* __restrict__ Wi,
                       float* __restrict__ P) {
    constexpr int LDT = 40;   // bf16 elems/row: 80B rows -> 16B-aligned b128, conflict-free banks
    __shared__ __align__(16) __bf16 sA[2][128 * LDT];
    __shared__ __align__(16) __bf16 sB[2][256 * LDT];
    const int t    = threadIdx.x;
    const int wv   = t >> 5;
    const int lane = t & 31;
    const int kh   = lane >> 4;       // lane half selects K sub-range (WMMA layout)
    const int ln   = lane & 15;
    const int mBase = blockIdx.x * 128;

    const v8f vzero = {0.f,0.f,0.f,0.f,0.f,0.f,0.f,0.f};
    v8f acc[16];
#pragma unroll
    for (int i = 0; i < 16; ++i) acc[i] = vzero;

    float4 aReg[4];               // A tile: 128x32 f32, 4 x float4 per thread
    float  bReg[8][4];            // B tile: 256x32 f32 (row stride 1025 -> 4B aligned, scalar loads)

    auto loadTile = [&](int k0) {
#pragma unroll
        for (int i = 0; i < 4; ++i) {
            const int e = t + i * 256, row = e >> 3, c4 = e & 7;
            aReg[i] = *(const float4*)(x + (size_t)(mBase + row) * INPUT_D + k0 + c4 * 4);
        }
#pragma unroll
        for (int i = 0; i < 8; ++i) {
            const int e = t + i * 256, n = e >> 3, c4 = e & 7;
            if (n < NINT) {
                const float* s = Wi + (size_t)n * WIROW + 1 + k0 + c4 * 4;
                bReg[i][0] = s[0]; bReg[i][1] = s[1]; bReg[i][2] = s[2]; bReg[i][3] = s[3];
            } else {
                bReg[i][0] = bReg[i][1] = bReg[i][2] = bReg[i][3] = 0.f;
            }
        }
    };
    auto storeTile = [&](int buf) {
#pragma unroll
        for (int i = 0; i < 4; ++i) {
            const int e = t + i * 256, row = e >> 3, c4 = e & 7;
            __bf16* d = sA[buf] + row * LDT + c4 * 4;
            d[0] = (__bf16)aReg[i].x; d[1] = (__bf16)aReg[i].y;
            d[2] = (__bf16)aReg[i].z; d[3] = (__bf16)aReg[i].w;
        }
#pragma unroll
        for (int i = 0; i < 8; ++i) {
            const int e = t + i * 256, n = e >> 3, c4 = e & 7;
            __bf16* d = sB[buf] + n * LDT + c4 * 4;
            d[0] = (__bf16)bReg[i][0]; d[1] = (__bf16)bReg[i][1];
            d[2] = (__bf16)bReg[i][2]; d[3] = (__bf16)bReg[i][3];
        }
    };
    auto computeTile = [&](int buf) {
        // A frag: row = wv*16+ln, K pairs {kh*8..+8} and {16+kh*8..+8}
        const __bf16* ap = sA[buf] + (wv * 16 + ln) * LDT + kh * 8;
        const v16bf afrag = cat8(*(const v8bf*)ap, *(const v8bf*)(ap + 16));
#pragma unroll
        for (int tn = 0; tn < 16; ++tn) {
            const __bf16* bp = sB[buf] + (tn * 16 + ln) * LDT + kh * 8;
            const v16bf bfrag = cat8(*(const v8bf*)bp, *(const v8bf*)(bp + 16));
            acc[tn] = __builtin_amdgcn_wmma_f32_16x16x32_bf16(
                false, afrag, false, bfrag, (short)0, acc[tn], false, false);
        }
    };

    loadTile(0);
    storeTile(0);
    __syncthreads();
    for (int kk = 0; kk < 32; ++kk) {
        const int cur = kk & 1;
        if (kk + 1 < 32) loadTile((kk + 1) * 32);   // global loads in flight during WMMAs
        computeTile(cur);
        if (kk + 1 < 32) {
            storeTile(cur ^ 1);                     // s_wait_loadcnt lands here, not before compute
            __syncthreads();
        }
    }

    // epilogue: +bias, sigmoid, store (C/D layout: row = r + 8*kh, col = ln)
#pragma unroll
    for (int tn = 0; tn < 16; ++tn) {
        const int n = tn * 16 + ln;
        const float bias = (n < NINT) ? Wi[(size_t)n * WIROW] : 0.f;
#pragma unroll
        for (int r = 0; r < 8; ++r) {
            const int row = mBase + wv * 16 + kh * 8 + r;
            const float z = acc[tn][r] + bias;
            const float pr = 1.0f / (1.0f + __expf(-z));
            if (n < NINT) P[(size_t)row * NINTP + n] = pr;
        }
    }
}

// ---------------------------------------------------------------------------
// Kernel 2: per-row tree walk. One wave per batch row (8 rows/block).
// Writes muLeaf (bf16, feeds GEMM2) and deterministic per-block leaf sums.
// ---------------------------------------------------------------------------
__global__ __launch_bounds__(256, 1)
void sdt_tree(const float* __restrict__ P,
              __bf16* __restrict__ muLeaf,
              float* __restrict__ blockSums) {
    __shared__ float pSh[8 * 256];
    __shared__ float ls[8 * 256];
    const int t = threadIdx.x;
    const int rowBase = blockIdx.x * 8;
#pragma unroll
    for (int i = 0; i < 8; ++i)
        pSh[i * 256 + t] = P[(size_t)(rowBase + i) * NINTP + t];
    __syncthreads();

    const int wv = t >> 5, lane = t & 31;
    const float* pr = pSh + wv * 256;
    const size_t row = rowBase + wv;
#pragma unroll
    for (int jj = 0; jj < 8; ++jj) {
        const int leaf = jj * 32 + lane;            // coalesced stores
        float mu = 1.f;
        int idx = 0;
#pragma unroll
        for (int d = 0; d < 8; ++d) {               // MSB-first: 0 -> p (left), 1 -> 1-p
            const int node = (1 << d) - 1 + idx;
            const int bit  = (leaf >> (7 - d)) & 1;
            const float pv = pr[node];
            mu *= bit ? (1.f - pv) : pv;
            idx = 2 * idx + bit;
        }
        muLeaf[row * NLEAF + leaf] = (__bf16)mu;
        ls[wv * 256 + leaf] = mu;
    }
    __syncthreads();
    float s = 0.f;                                  // fixed-order intra-block reduction
#pragma unroll
    for (int w = 0; w < 8; ++w) s += ls[w * 256 + t];
    blockSums[(size_t)blockIdx.x * 256 + t] = s;
}

// Fixed-order inter-block reduction -> leafS[256]  (deterministic, no atomics)
__global__ void sdt_reduce(const float* __restrict__ blockSums,
                           float* __restrict__ leafS) {
    const int t = threadIdx.x;
    float s = 0.f;
    for (int b = 0; b < BATCH / 8; ++b)
        s += blockSums[(size_t)b * 256 + t];
    leafS[t] = s;
}

// Penalty from leaf sums: S[parent]=S[l]+S[r]; alpha = S[child]/S[parent].
__global__ void sdt_penalty(const float* __restrict__ leafS,
                            float* __restrict__ outPen) {
    if (threadIdx.x != 0 || blockIdx.x != 0) return;
    float S[511];
    for (int i = 0; i < 256; ++i) S[255 + i] = leafS[i];
    for (int n = 254; n >= 0; --n) S[n] = S[2 * n + 1] + S[2 * n + 2];
    float acc = 0.f;
    for (int l = 0; l < 8; ++l) {
        const float coeff = 1e-3f * exp2f((float)-l);
        const int beg = (2 << l) - 1, cnt = 2 << l;   // children at depth l+1
        for (int c = beg; c < beg + cnt; ++c) {
            const float alpha = S[c] / S[(c - 1) >> 1];
            acc += coeff * (__logf(alpha) + log1pf(-alpha));
        }
    }
    *outPen = -0.5f * acc;
}

// ---------------------------------------------------------------------------
// Kernel 5: Y = muLeaf(bf16) @ W_leaf^T. BM=128 x BN=64, K=256 (8 steps),
// same double-buffered pipeline; 4 WMMAs per wave per step.
// ---------------------------------------------------------------------------
__global__ __launch_bounds__(256, 1)
void sdt_gemm2(const __bf16* __restrict__ muLeaf,
               const float* __restrict__ Wl,
               float* __restrict__ Y) {
    constexpr int LDT = 40;
    __shared__ __align__(16) __bf16 sA[2][128 * LDT];
    __shared__ __align__(16) __bf16 sB[2][64 * LDT];
    const int t    = threadIdx.x;
    const int wv   = t >> 5;
    const int lane = t & 31;
    const int kh   = lane >> 4;
    const int ln   = lane & 15;
    const int mBase = blockIdx.x * 128;
    const int nBase = blockIdx.y * 64;

    const v8f vzero = {0.f,0.f,0.f,0.f,0.f,0.f,0.f,0.f};
    v8f acc[4];
#pragma unroll
    for (int i = 0; i < 4; ++i) acc[i] = vzero;

    uint4  aReg[2];   // A tile: 128x32 bf16 already packed -> 16B copies
    float4 bReg[2];   // B tile: 64x32 f32 (W_leaf rows 1024B aligned -> float4)

    auto loadTile = [&](int k0) {
#pragma unroll
        for (int i = 0; i < 2; ++i) {
            const int e = t + i * 256, row = e >> 2, seg = e & 3;
            aReg[i] = *(const uint4*)(muLeaf + (size_t)(mBase + row) * NLEAF + k0 + seg * 8);
        }
#pragma unroll
        for (int i = 0; i < 2; ++i) {
            const int e = t + i * 256, nl = e >> 3, c4 = e & 7;
            const int n = nBase + nl;
            bReg[i] = (n < NOUT) ? *(const float4*)(Wl + (size_t)n * NLEAF + k0 + c4 * 4)
                                 : make_float4(0.f, 0.f, 0.f, 0.f);
        }
    };
    auto storeTile = [&](int buf) {
#pragma unroll
        for (int i = 0; i < 2; ++i) {
            const int e = t + i * 256, row = e >> 2, seg = e & 3;
            *(uint4*)(sA[buf] + row * LDT + seg * 8) = aReg[i];
        }
#pragma unroll
        for (int i = 0; i < 2; ++i) {
            const int e = t + i * 256, nl = e >> 3, c4 = e & 7;
            __bf16* d = sB[buf] + nl * LDT + c4 * 4;
            d[0] = (__bf16)bReg[i].x; d[1] = (__bf16)bReg[i].y;
            d[2] = (__bf16)bReg[i].z; d[3] = (__bf16)bReg[i].w;
        }
    };
    auto computeTile = [&](int buf) {
        const __bf16* ap = sA[buf] + (wv * 16 + ln) * LDT + kh * 8;
        const v16bf afrag = cat8(*(const v8bf*)ap, *(const v8bf*)(ap + 16));
#pragma unroll
        for (int tn = 0; tn < 4; ++tn) {
            const __bf16* bp = sB[buf] + (tn * 16 + ln) * LDT + kh * 8;
            const v16bf bfrag = cat8(*(const v8bf*)bp, *(const v8bf*)(bp + 16));
            acc[tn] = __builtin_amdgcn_wmma_f32_16x16x32_bf16(
                false, afrag, false, bfrag, (short)0, acc[tn], false, false);
        }
    };

    loadTile(0);
    storeTile(0);
    __syncthreads();
    for (int kk = 0; kk < 8; ++kk) {
        const int cur = kk & 1;
        if (kk + 1 < 8) loadTile((kk + 1) * 32);
        computeTile(cur);
        if (kk + 1 < 8) {
            storeTile(cur ^ 1);
            __syncthreads();
        }
    }

#pragma unroll
    for (int tn = 0; tn < 4; ++tn) {
        const int n = nBase + tn * 16 + ln;
        if (n < NOUT) {
#pragma unroll
            for (int r = 0; r < 8; ++r) {
                const int row = mBase + wv * 16 + kh * 8 + r;
                Y[(size_t)row * NOUT + n] = acc[tn][r];
            }
        }
    }
}

// ---------------------------------------------------------------------------
extern "C" void kernel_launch(void* const* d_in, const int* in_sizes, int n_in,
                              void* d_out, int out_size, void* d_ws, size_t ws_size,
                              hipStream_t stream) {
    (void)in_sizes; (void)n_in; (void)out_size; (void)ws_size;
    const float* x  = (const float*)d_in[0];   // [32768,1024]
    const float* Wi = (const float*)d_in[1];   // [255,1025]
    const float* Wl = (const float*)d_in[2];   // [1000,256]
    float* out = (float*)d_out;                // y[32768*1000] then penalty scalar

    char* ws = (char*)d_ws;
    float*  P         = (float*)(ws);                                  // 33,554,432 B
    __bf16* muLeaf    = (__bf16*)(ws + (size_t)33554432);              // 16,777,216 B
    float*  blockSums = (float*)(ws + (size_t)33554432 + 16777216);    //  4,194,304 B
    float*  leafS     = (float*)(ws + (size_t)33554432 + 16777216 + 4194304); // 1,024 B

    sdt_gemm1_sigmoid<<<dim3(BATCH / 128), dim3(256), 0, stream>>>(x, Wi, P);
    sdt_tree<<<dim3(BATCH / 8), dim3(256), 0, stream>>>(P, muLeaf, blockSums);
    sdt_reduce<<<dim3(1), dim3(256), 0, stream>>>(blockSums, leafS);
    sdt_penalty<<<dim3(1), dim3(1), 0, stream>>>(leafS, out + (size_t)BATCH * NOUT);
    sdt_gemm2<<<dim3(BATCH / 128, 16), dim3(256), 0, stream>>>(muLeaf, Wl, out);
}
